// DeepIntersetNetwork_5626407157924
// MI455X (gfx1250) — compile-verified, matrix operand order checked by reading, not run on hardware
//
#include <hip/hip_runtime.h>
#include <hip/hip_bf16.h>

#define B_      512
#define S_      100
#define F_      8
#define FP_     4
#define E_      64
#define V_ITEM  100000
#define V_PROF  10000
#define CTX_    64
#define FE      (F_*E_)                 // 512
#define ATT_IN  (4*FE)                  // 2048
#define TOWER_IN (FE+FE+FP_*E_+CTX_)    // 1344
#define LN_EPS  1e-3f

typedef __bf16 bf16_t;
typedef __attribute__((ext_vector_type(16))) __bf16 v16bf;
typedef __attribute__((ext_vector_type(8)))  __bf16 v8bf;
typedef __attribute__((ext_vector_type(8)))  float  v8f;

// ---------------------------------------------------------------------------
// Convert f32 weight (K x N, row-major) -> bf16 transposed (N x K, row-major)
// so the WMMA B operand can be loaded contiguously along K.
// ---------------------------------------------------------------------------
__global__ void k_transpose_bf16(const float* __restrict__ W,
                                 bf16_t* __restrict__ Wt, int K, int N) {
    int idx = blockIdx.x * blockDim.x + threadIdx.x;
    if (idx >= K * N) return;
    int k = idx / N, n = idx - k * N;
    Wt[(size_t)n * K + k] = (bf16_t)W[idx];
}

// ---------------------------------------------------------------------------
// Embedding gather + DIN attention + attention-weighted mean pool.
// One block per batch row; each thread owns 2 of the 512 feature lanes.
// Attention logit decomposition avoids materializing the (B,S,2048) tensor:
//   feat . att_w[s] = sum_k  u*w0 + t*w1 + u*t*w2 + (u+t)*w3
// This kernel is the roofline bottleneck (~105MB of random 256B embedding
// rows); prefetch the next step's attention weights across the barrier-
// serialized reduction (lowers to global_prefetch_b8 on gfx1250).
// ---------------------------------------------------------------------------
__global__ void __launch_bounds__(256) k_gather_att_pool(
    const int* __restrict__ target_ad, const int* __restrict__ ubs,
    const int* __restrict__ prof_idx,  const float* __restrict__ ctx,
    const float* __restrict__ item_emb, const float* __restrict__ prof_emb,
    const float* __restrict__ att_w,    const float* __restrict__ att_b,
    bf16_t* __restrict__ X0)
{
    __shared__ int   sIdx[F_];
    __shared__ float red[8];
    __shared__ float sAtt;

    int b   = blockIdx.x;
    int tid = threadIdx.x;
    int k0 = tid * 2, k1 = k0 + 1;
    int f0 = k0 >> 6, e0 = k0 & 63;
    int f1 = k1 >> 6, e1 = k1 & 63;

    float t0 = item_emb[(size_t)f0 * V_ITEM * E_ + (size_t)target_ad[b*F_ + f0] * E_ + e0];
    float t1 = item_emb[(size_t)f1 * V_ITEM * E_ + (size_t)target_ad[b*F_ + f1] * E_ + e1];
    float p0 = 0.f, p1 = 0.f;

    for (int s = 0; s < S_; ++s) {
        __syncthreads();                       // protect shared state from prev iter
        if (tid < F_) sIdx[tid] = ubs[(b * S_ + s) * F_ + tid];
        __syncthreads();

        float u0 = item_emb[(size_t)f0 * V_ITEM * E_ + (size_t)sIdx[f0] * E_ + e0];
        float u1 = item_emb[(size_t)f1 * V_ITEM * E_ + (size_t)sIdx[f1] * E_ + e1];

        const float* aw = att_w + (size_t)s * ATT_IN;
        if (s + 1 < S_) {                      // hide latency across the barriers
            __builtin_prefetch(aw + ATT_IN + k0, 0, 1);
            __builtin_prefetch(aw + ATT_IN + 2*FE + k0, 0, 1);
            if (tid < F_) __builtin_prefetch(&ubs[(b * S_ + s + 1) * F_ + tid], 0, 1);
        }
        float part = u0*aw[k0] + t0*aw[FE+k0] + u0*t0*aw[2*FE+k0] + (u0+t0)*aw[3*FE+k0]
                   + u1*aw[k1] + t1*aw[FE+k1] + u1*t1*aw[2*FE+k1] + (u1+t1)*aw[3*FE+k1];

        for (int off = 16; off > 0; off >>= 1) part += __shfl_xor(part, off, 32);
        if ((tid & 31) == 0) red[tid >> 5] = part;
        __syncthreads();
        if (tid == 0) {
            float acc = att_b[s];
            #pragma unroll
            for (int i = 0; i < 8; ++i) acc += red[i];
            sAtt = 1.f / (1.f + __expf(-acc));
        }
        __syncthreads();
        float a = sAtt;
        p0 += u0 * a;
        p1 += u1 * a;
    }

    const float invS = 1.f / (float)S_;
    bf16_t* row = X0 + (size_t)b * TOWER_IN;
    row[k0]      = (bf16_t)(p0 * invS);
    row[k1]      = (bf16_t)(p1 * invS);
    row[FE + k0] = (bf16_t)t0;
    row[FE + k1] = (bf16_t)t1;
    if (tid < 128) {                           // profile embeddings: 256 lanes
        int kp0 = tid * 2, kp1 = kp0 + 1;
        int fp0 = kp0 >> 6, ep0 = kp0 & 63;
        int fp1 = kp1 >> 6, ep1 = kp1 & 63;
        row[2*FE + kp0] = (bf16_t)prof_emb[(size_t)fp0 * V_PROF * E_ + (size_t)prof_idx[b*FP_ + fp0] * E_ + ep0];
        row[2*FE + kp1] = (bf16_t)prof_emb[(size_t)fp1 * V_PROF * E_ + (size_t)prof_idx[b*FP_ + fp1] * E_ + ep1];
    }
    if (tid < CTX_) row[2*FE + FP_*E_ + tid] = (bf16_t)ctx[b * CTX_ + tid];
}

// ---------------------------------------------------------------------------
// bf16 WMMA GEMM: Y[M,N] = A[M,K] @ B[K,N] + bias, with Bt = B^T (N x K).
// Each wave computes a 16x64 output slab: one shared A fragment feeds 4
// accumulators / 4 v_wmma_f32_16x16x32_bf16 per K-step (2 A + 8 B b128 loads
// per 4 WMMAs -> 2.5 loads/WMMA, 4x less A traffic than one-tile waves).
// 16-bit A VGPR layout (ISA 7.12.2): lanes 0-15 hold K=0..7 & 16..23 of row M,
// lanes 16-31 hold K=8..15 & 24..31 -> two contiguous 16B loads per operand.
// ---------------------------------------------------------------------------
__global__ void __launch_bounds__(128) k_gemm_bf16_wmma(
    const bf16_t* __restrict__ A,    // M x K row-major
    const bf16_t* __restrict__ Bt,   // N x K row-major
    const float*  __restrict__ bias, // N
    float* __restrict__ Y,           // M x N
    int M, int N, int K)
{
    int lane = threadIdx.x;          // wave32 lane
    int hl   = lane >> 4;            // half-wave select
    int l16  = lane & 15;
    int nBase = (blockIdx.x * 4 + threadIdx.y) * 64;   // 4 N-tiles per wave
    int mBase = blockIdx.y * 16;

    const bf16_t* arow = A  + (size_t)(mBase + l16) * K + 8 * hl;
    const bf16_t* brow = Bt + (size_t)(nBase + l16) * K + 8 * hl;

    v8f acc0 = {}, acc1 = {}, acc2 = {}, acc3 = {};
    for (int kb = 0; kb < K; kb += 32) {
        v16bf a, b0, b1, b2, b3;
        ((v8bf*)&a)[0]  = *(const v8bf*)(arow + kb);            // K = kb+8h..+7
        ((v8bf*)&a)[1]  = *(const v8bf*)(arow + kb + 16);       // K = kb+16+8h..
        ((v8bf*)&b0)[0] = *(const v8bf*)(brow + kb);
        ((v8bf*)&b0)[1] = *(const v8bf*)(brow + kb + 16);
        ((v8bf*)&b1)[0] = *(const v8bf*)(brow + (size_t)16*K + kb);
        ((v8bf*)&b1)[1] = *(const v8bf*)(brow + (size_t)16*K + kb + 16);
        ((v8bf*)&b2)[0] = *(const v8bf*)(brow + (size_t)32*K + kb);
        ((v8bf*)&b2)[1] = *(const v8bf*)(brow + (size_t)32*K + kb + 16);
        ((v8bf*)&b3)[0] = *(const v8bf*)(brow + (size_t)48*K + kb);
        ((v8bf*)&b3)[1] = *(const v8bf*)(brow + (size_t)48*K + kb + 16);
        acc0 = __builtin_amdgcn_wmma_f32_16x16x32_bf16(false, a, false, b0, (short)0, acc0, false, false);
        acc1 = __builtin_amdgcn_wmma_f32_16x16x32_bf16(false, a, false, b1, (short)0, acc1, false, false);
        acc2 = __builtin_amdgcn_wmma_f32_16x16x32_bf16(false, a, false, b2, (short)0, acc2, false, false);
        acc3 = __builtin_amdgcn_wmma_f32_16x16x32_bf16(false, a, false, b3, (short)0, acc3, false, false);
    }

    // D layout: VGPR v -> row v + 8*hl, col l16 (per 16-wide tile)
    #pragma unroll
    for (int t = 0; t < 4; ++t) {
        int col = nBase + 16 * t + l16;
        float bs = bias[col];
        const v8f& acc = (t == 0) ? acc0 : (t == 1) ? acc1 : (t == 2) ? acc2 : acc3;
        #pragma unroll
        for (int v = 0; v < 8; ++v) {
            int m = mBase + v + 8 * hl;
            Y[(size_t)m * N + col] = acc[v] + bs;
        }
    }
}

// ---------------------------------------------------------------------------
// Row-wise LayerNorm (+ReLU) matching reference var = E[x^2] - mu^2, bf16 out.
// ---------------------------------------------------------------------------
__global__ void __launch_bounds__(256) k_ln_relu_bf16(
    const float* __restrict__ Y, const float* __restrict__ g,
    const float* __restrict__ be, bf16_t* __restrict__ H, int N)
{
    __shared__ float rs[8], rq[8];
    int b = blockIdx.x, tid = threadIdx.x;
    const float* row = Y + (size_t)b * N;

    float s = 0.f, q = 0.f;
    for (int i = tid; i < N; i += 256) { float x = row[i]; s += x; q += x * x; }
    for (int off = 16; off > 0; off >>= 1) {
        s += __shfl_xor(s, off, 32);
        q += __shfl_xor(q, off, 32);
    }
    if ((tid & 31) == 0) { rs[tid >> 5] = s; rq[tid >> 5] = q; }
    __syncthreads();
    if (tid == 0) {
        float ts = 0.f, tq = 0.f;
        #pragma unroll
        for (int i = 0; i < 8; ++i) { ts += rs[i]; tq += rq[i]; }
        float mu = ts / (float)N;
        rs[0] = mu;
        rq[0] = rsqrtf(tq / (float)N - mu * mu + LN_EPS);
    }
    __syncthreads();
    float mu = rs[0], rstd = rq[0];
    bf16_t* hrow = H + (size_t)b * N;
    for (int i = tid; i < N; i += 256) {
        float y = (row[i] - mu) * rstd * g[i] + be[i];
        hrow[i] = (bf16_t)fmaxf(y, 0.f);
    }
}

// ---------------------------------------------------------------------------
// Final 256 -> 1 projection + sigmoid, f32 output.
// ---------------------------------------------------------------------------
__global__ void __launch_bounds__(256) k_final_dot(
    const bf16_t* __restrict__ H, const float* __restrict__ w4,
    const float* __restrict__ b4, float* __restrict__ out)
{
    __shared__ float red[8];
    int b = blockIdx.x, tid = threadIdx.x;
    float v = (float)H[(size_t)b * 256 + tid] * w4[tid];
    for (int off = 16; off > 0; off >>= 1) v += __shfl_xor(v, off, 32);
    if ((tid & 31) == 0) red[tid >> 5] = v;
    __syncthreads();
    if (tid == 0) {
        float acc = b4[0];
        #pragma unroll
        for (int i = 0; i < 8; ++i) acc += red[i];
        out[b] = 1.f / (1.f + __expf(-acc));
    }
}

// ---------------------------------------------------------------------------
extern "C" void kernel_launch(void* const* d_in, const int* in_sizes, int n_in,
                              void* d_out, int out_size, void* d_ws, size_t ws_size,
                              hipStream_t stream)
{
    (void)in_sizes; (void)n_in; (void)out_size; (void)ws_size;
    const int*   target_ad = (const int*)d_in[0];
    const int*   ubs       = (const int*)d_in[1];
    const int*   prof_idx  = (const int*)d_in[2];
    const float* ctx       = (const float*)d_in[3];
    const float* item_emb  = (const float*)d_in[4];
    const float* prof_emb  = (const float*)d_in[5];
    const float* att_w     = (const float*)d_in[6];
    const float* att_b     = (const float*)d_in[7];
    const float* w1  = (const float*)d_in[8];  const float* b1  = (const float*)d_in[9];
    const float* g1  = (const float*)d_in[10]; const float* be1 = (const float*)d_in[11];
    const float* w2  = (const float*)d_in[12]; const float* b2  = (const float*)d_in[13];
    const float* g2  = (const float*)d_in[14]; const float* be2 = (const float*)d_in[15];
    const float* w3  = (const float*)d_in[16]; const float* b3  = (const float*)d_in[17];
    const float* g3  = (const float*)d_in[18]; const float* be3 = (const float*)d_in[19];
    const float* w4  = (const float*)d_in[20]; const float* b4  = (const float*)d_in[21];
    float* out = (float*)d_out;

    // workspace carve-out (~9.4 MB total), 256B aligned
    char* ws = (char*)d_ws;
    size_t off = 0;
    auto carve = [&](size_t bytes) -> char* {
        char* p = ws + off;
        off = (off + bytes + 255) & ~(size_t)255;
        return p;
    };
    bf16_t* X0   = (bf16_t*)carve((size_t)B_ * TOWER_IN * 2);
    bf16_t* W1t  = (bf16_t*)carve((size_t)1024 * TOWER_IN * 2);
    bf16_t* W2t  = (bf16_t*)carve((size_t)512 * 1024 * 2);
    bf16_t* W3t  = (bf16_t*)carve((size_t)256 * 512 * 2);
    float*  Ybuf = (float*) carve((size_t)B_ * 1024 * 4);
    bf16_t* H1   = (bf16_t*)carve((size_t)B_ * 1024 * 2);
    bf16_t* H2   = (bf16_t*)carve((size_t)B_ * 512 * 2);
    bf16_t* H3   = (bf16_t*)carve((size_t)B_ * 256 * 2);

    // weights -> bf16 transposed
    { int n = TOWER_IN * 1024; k_transpose_bf16<<<(n + 255) / 256, 256, 0, stream>>>(w1, W1t, TOWER_IN, 1024); }
    { int n = 1024 * 512;      k_transpose_bf16<<<(n + 255) / 256, 256, 0, stream>>>(w2, W2t, 1024, 512); }
    { int n = 512 * 256;       k_transpose_bf16<<<(n + 255) / 256, 256, 0, stream>>>(w3, W3t, 512, 256); }

    // gather + attention + pool -> tower input (bf16)
    k_gather_att_pool<<<B_, 256, 0, stream>>>(target_ad, ubs, prof_idx, ctx,
                                              item_emb, prof_emb, att_w, att_b, X0);

    // MLP tower: WMMA GEMM (16x64 slab per wave) + fused bias, then LN+ReLU
    k_gemm_bf16_wmma<<<dim3(1024 / 256, B_ / 16), dim3(32, 4), 0, stream>>>(X0, W1t, b1, Ybuf, B_, 1024, TOWER_IN);
    k_ln_relu_bf16<<<B_, 256, 0, stream>>>(Ybuf, g1, be1, H1, 1024);

    k_gemm_bf16_wmma<<<dim3(512 / 256, B_ / 16), dim3(32, 4), 0, stream>>>(H1, W2t, b2, Ybuf, B_, 512, 1024);
    k_ln_relu_bf16<<<B_, 256, 0, stream>>>(Ybuf, g2, be2, H2, 512);

    k_gemm_bf16_wmma<<<dim3(256 / 256, B_ / 16), dim3(32, 4), 0, stream>>>(H2, W3t, b3, Ybuf, B_, 256, 512);
    k_ln_relu_bf16<<<B_, 256, 0, stream>>>(Ybuf, g3, be3, H3, 256);

    k_final_dot<<<B_, 256, 0, stream>>>(H3, w4, b4, out);
}
